// VectorQuantizerEMA_84799834292275
// MI455X (gfx1250) — matrix operand order repaired and checked by reading, not compiled, not run on hardware
//
#include <hip/hip_runtime.h>

// ---------------------------------------------------------------------------
// VQ-VAE EMA vector quantizer for MI455X (gfx1250, wave32, WMMA).
// Distance GEMM (16384x4096x512) on v_wmma_f32_16x16x32_bf16 with the B
// (codebook) tiles staged through LDS via global_load_async_to_lds_b128,
// consumed through an 8-deep register ring of ds_load_b128 fragments.
// ---------------------------------------------------------------------------

typedef __attribute__((ext_vector_type(16))) __bf16 v16bf;
typedef __attribute__((ext_vector_type(8)))  __bf16 v8bf;
typedef __attribute__((ext_vector_type(8)))  float  v8f;
typedef __attribute__((ext_vector_type(4)))  float  f4;

#define SS 256
#define BB 64
#define DD 512
#define KK 4096
#define NN (SS * BB)                 // 16384 rows

#define DECAY_C   0.99f
#define ONEMD_C   0.01f
#define EPS_C     1e-05f
#define COMMIT_C  0.25f

// LDS B-tile geometry: 16 rows x 512 bf16, row padded 1024B -> 1040B so the
// 32 ds_load_b128 requests/instr spread 2-per-bank (minimum service time).
#define ROWB      1040               // bytes per padded LDS row
#define ROWE      520                // ushort elements per padded LDS row
#define BUFB      (16 * ROWB)        // 16640 bytes per buffer
#define BUFE      (16 * ROWE)        // 8320 elements per buffer

// --- d_out layout (floats, reference tuple order) --------------------------
#define O_LOSS  ((size_t)0)
#define O_Q     ((size_t)1)
#define O_ENC   (O_Q   + (size_t)NN * DD)        // 8388609
#define O_IDX   (O_ENC + (size_t)NN * KK)
#define O_W     (O_IDX + (size_t)NN)
#define O_CS    (O_W   + (size_t)KK * DD)
#define O_EMAW  (O_CS  + (size_t)KK)

// --- workspace layout (bytes) ----------------------------------------------
#define WS_XB     ((size_t)0)                        // NN*DD bf16 = 16 MB
#define WS_WB     (WS_XB + (size_t)NN * DD * 2)      // KK*DD bf16 =  4 MB
#define WS_DW     (WS_WB + (size_t)KK * DD * 2)      // KK*DD f32  =  8 MB
#define WS_ENORM  (WS_DW + (size_t)KK * DD * 4)      // KK f32
#define WS_CNT    (WS_ENORM + (size_t)KK * 4)        // KK f32
#define WS_NCS    (WS_CNT   + (size_t)KK * 4)        // KK f32
#define WS_SCAL   (WS_NCS   + (size_t)KK * 4)        // [0]=lossAcc [1]=nSum

__device__ __forceinline__ unsigned short f2bf(float f) {
  unsigned int u = __float_as_uint(f);
  unsigned int r = u + 0x7FFFu + ((u >> 16) & 1u);   // round-to-nearest-even
  return (unsigned short)(r >> 16);
}

// Raw workgroup split-barrier.  Cross-wave LDS visibility of the staged tile
// is already enforced by each wave's s_wait_asynccnt before signaling, and
// the consuming ds_loads complete before the next signal (they feed WMMAs),
// so the full VMEM drain __syncthreads() performs is unnecessary.
__device__ __forceinline__ void wg_barrier() {
  asm volatile("s_barrier_signal -1\n\t"
               "s_barrier_wait -1" ::: "memory");
}

// ---------------------------------------------------------------------------
__global__ void zero_kernel(float* __restrict__ p, size_t n) {
  size_t i  = (size_t)blockIdx.x * blockDim.x + threadIdx.x;
  size_t st = (size_t)gridDim.x * blockDim.x;
  for (; i < n; i += st) __builtin_nontemporal_store(0.0f, p + i);
}

// X -> bf16
__global__ void xconv_kernel(const float* __restrict__ X,
                             unsigned short* __restrict__ Xb) {
  size_t i  = (size_t)blockIdx.x * blockDim.x + threadIdx.x;
  size_t st = (size_t)gridDim.x * blockDim.x;
  size_t n  = (size_t)NN * DD;
  for (; i < n; i += st) Xb[i] = f2bf(X[i]);
}

// W -> bf16 and ||e_k||^2 (one block of 128 threads per codebook row)
__global__ __launch_bounds__(128) void wprep_kernel(const float* __restrict__ W,
                                                    unsigned short* __restrict__ Wb,
                                                    float* __restrict__ enorm) {
  const int k = blockIdx.x;
  const int t = threadIdx.x;                 // 0..127, covers 512 floats as f4
  const float* wr = W + (size_t)k * DD + t * 4;
  f4 v = *(const f4*)wr;
  unsigned short* wb = Wb + (size_t)k * DD + t * 4;
  wb[0] = f2bf(v[0]); wb[1] = f2bf(v[1]); wb[2] = f2bf(v[2]); wb[3] = f2bf(v[3]);
  float s = v[0]*v[0] + v[1]*v[1] + v[2]*v[2] + v[3]*v[3];
  #pragma unroll
  for (int m = 1; m < 32; m <<= 1) s += __shfl_xor(s, m, 32);
  __shared__ float red[4];
  if ((t & 31) == 0) red[t >> 5] = s;
  __syncthreads();
  if (t == 0) enorm[k] = red[0] + red[1] + red[2] + red[3];
}

// ---------------------------------------------------------------------------
// Stage one 16-code x 512-bf16 B tile (16 KB) into LDS asynchronously.
// 256 threads x 4 chunks of 16 B.  The instruction offset is added to BOTH
// the LDS and the global address (CDNA5 ISA), so one
// address pair serves all 4 chunks of a row segment.
__device__ __forceinline__ void stage_tile(const unsigned short* Wb, int n0,
                                           unsigned ldsBase, int tid) {
  const int r  = tid >> 4;                       // codebook row within tile
  const int cb = tid & 15;                       // 16B chunk within row
  unsigned long long g =
      (unsigned long long)(size_t)(Wb + (size_t)(n0 + r) * DD) + 16u * (unsigned)cb;
  unsigned l = ldsBase + (unsigned)(ROWB * r + 16 * cb);
  asm volatile(
      "global_load_async_to_lds_b128 %0, %1, off\n\t"
      "global_load_async_to_lds_b128 %0, %1, off offset:256\n\t"
      "global_load_async_to_lds_b128 %0, %1, off offset:512\n\t"
      "global_load_async_to_lds_b128 %0, %1, off offset:768"
      :: "v"(l), "v"(g) : "memory");
}

// ---------------------------------------------------------------------------
// Main kernel: one wave owns a 16-row M tile (A resident in 128 VGPRs); the
// 8 waves of a block share each B tile through triple-buffered LDS.
// ---------------------------------------------------------------------------
__global__ __launch_bounds__(256) void vq_argmin_kernel(
    const unsigned short* __restrict__ Xb, const unsigned short* __restrict__ Wb,
    const float* __restrict__ enorm,
    const float* __restrict__ Xf, const float* __restrict__ Wf,
    float* __restrict__ out_q, float* __restrict__ out_enc,
    float* __restrict__ out_idx,
    float* __restrict__ dw, float* __restrict__ counts,
    float* __restrict__ lossAcc) {
  __shared__ unsigned short Bt[3 * BUFE];        // 49,920 B triple buffer

  const int tid  = (int)threadIdx.x;
  const int lane = tid & 31;
  const int wid  = ((int)blockIdx.x * 256 + tid) >> 5;
  const int m0   = wid * 16;
  const int nl   = lane & 15;     // A: row within M tile ; B/C: column N
  const int kh   = lane >> 4;     // K-half selector (ISA 16-bit layouts)

  const __bf16* XB = (const __bf16*)Xb;
  const unsigned ldsBase = (unsigned)(size_t)&Bt[0];   // low 32 bits = LDS offset

  // ---- preload A fragments (rows m0..m0+15, all D=512, 16 chunks of K=32)
  v16bf afrag[16];
  {
    const __bf16* xrow = XB + (size_t)(m0 + nl) * DD;
    #pragma unroll
    for (int c = 0; c < 16; ++c) {
      // lanes 0-15: elems 0-7 = K 0..7, elems 8-15 = K 16..23  (kh = 0)
      // lanes16-31: elems 0-7 = K 8..15, elems 8-15 = K 24..31 (kh = 1)
      v8bf lo = *(const v8bf*)(xrow + 32 * c + kh * 8);
      v8bf hi = *(const v8bf*)(xrow + 32 * c + 16 + kh * 8);
      afrag[c] = __builtin_shufflevector(lo, hi, 0, 1, 2, 3, 4, 5, 6, 7,
                                                 8, 9, 10, 11, 12, 13, 14, 15);
    }
  }

  float bval[8];
  int   bidx[8];
  #pragma unroll
  for (int i = 0; i < 8; ++i) { bval[i] = 3.4e38f; bidx[i] = 0; }

  // ---- prologue: stage tile 0 into buffer 0
  stage_tile(Wb, 0, ldsBase, tid);

  // ---- sweep the 256 codebook tiles (triple-buffered LDS pipeline)
  int bcur = 0;
  #pragma unroll 1
  for (int tt = 0; tt < KK / 16; ++tt) {
    const int bnext = (bcur == 2) ? 0 : bcur + 1;
    if (tt + 1 < KK / 16) {
      stage_tile(Wb, (tt + 1) * 16, ldsBase + (unsigned)(bnext * BUFB), tid);
      asm volatile("s_wait_asynccnt 4" ::: "memory");   // tile tt complete
    } else {
      asm volatile("s_wait_asynccnt 0" ::: "memory");
    }
    wg_barrier();

    const int   n0 = tt * 16;
    const float en = enorm[n0 + nl];               // overlap with LDS chain

    // B 32x16 bf16: lane = N, elems 0-7 = K kh*8..kh*8+7, elems 8-15 = K 16+kh*8..16+kh*8+7.
    // 8-deep register ring of B fragments: slot c&7 is reloaded for frag c+8
    // BEFORE the WMMA that consumes its previous value, forcing distinct
    // registers so ~8 ds_load pairs stay in flight behind the WMMA chain.
    const __bf16* bb =
        (const __bf16*)&Bt[bcur * BUFE + nl * ROWE + kh * 8];
    auto ldfrag = [&](int c) -> v16bf {
      v8bf lo = *(const v8bf*)(bb + 32 * c);
      v8bf hi = *(const v8bf*)(bb + 32 * c + 16);
      return __builtin_shufflevector(lo, hi, 0, 1, 2, 3, 4, 5, 6, 7,
                                             8, 9, 10, 11, 12, 13, 14, 15);
    };
    v16bf bfrag[8];
    #pragma unroll
    for (int c = 0; c < 8; ++c) bfrag[c] = ldfrag(c);
    v8f acc = {};
    #pragma unroll
    for (int c = 0; c < 16; ++c) {
      v16bf bf = bfrag[c & 7];
      if (c + 8 < 16) bfrag[c & 7] = ldfrag(c + 8);  // issue ahead of the WMMA
      acc = __builtin_amdgcn_wmma_f32_16x16x32_bf16(
          false, afrag[c], false, bf, (short)0, acc, false, false);
    }

    const int cand = n0 + nl;
    #pragma unroll
    for (int i = 0; i < 8; ++i) {
      // C layout: lane = N column; VGPR i = row i (lanes 0-15) / 8+i (16-31)
      float d = __builtin_fmaf(-2.0f, acc[i], en);
      if (d < bval[i]) { bval[i] = d; bidx[i] = cand; }
    }
    bcur = bnext;
  }

  // ---- argmin across the 16 lanes of each half (ties -> smaller index)
  #pragma unroll
  for (int i = 0; i < 8; ++i) {
    float v  = bval[i];
    int   ix = bidx[i];
    #pragma unroll
    for (int m = 1; m < 16; m <<= 1) {     // stays within 16-lane half
      float ov = __shfl_xor(v, m, 32);
      int   oi = __shfl_xor(ix, m, 32);
      if (ov < v || (ov == v && oi < ix)) { v = ov; ix = oi; }
    }
    bidx[i] = ix;
  }

  // ---- cooperative per-row epilogue: gather code row, one-hot, dw, loss
  float lsum = 0.0f;
  #pragma unroll
  for (int r = 0; r < 16; ++r) {
    const int k = __shfl(bidx[r & 7], (r >> 3) * 16, 32);  // broadcast argmin
    const size_t m  = (size_t)(m0 + r);
    const float* wr = Wf + (size_t)k * DD;
    const float* xr = Xf + m * DD;
    float* qr  = out_q + m * DD;                // odd float offset: scalar stores
    float* dwr = dw + (size_t)k * DD;
    #pragma unroll
    for (int it = 0; it < 4; ++it) {
      const int d = lane * 4 + it * 128;
      f4 wv = *(const f4*)(wr + d);
      f4 xv = *(const f4*)(xr + d);
      qr[d + 0] = wv[0]; qr[d + 1] = wv[1]; qr[d + 2] = wv[2]; qr[d + 3] = wv[3];
      f4 df = wv - xv;
      lsum += df[0]*df[0] + df[1]*df[1] + df[2]*df[2] + df[3]*df[3];
      atomicAdd(dwr + d + 0, xv[0]);
      atomicAdd(dwr + d + 1, xv[1]);
      atomicAdd(dwr + d + 2, xv[2]);
      atomicAdd(dwr + d + 3, xv[3]);
    }
    if (lane == 0) {
      out_idx[m] = (float)k;
      out_enc[m * (size_t)KK + (size_t)k] = 1.0f;
      atomicAdd(counts + k, 1.0f);
    }
  }
  #pragma unroll
  for (int m = 1; m < 32; m <<= 1) lsum += __shfl_xor(lsum, m, 32);
  if (lane == 0) atomicAdd(lossAcc, lsum);
}

// ---------------------------------------------------------------------------
// EMA cluster size (raw) + global sum n
__global__ __launch_bounds__(256) void cs_kernel(const float* __restrict__ cs,
                                                 const float* __restrict__ counts,
                                                 float* __restrict__ ncsraw,
                                                 float* __restrict__ nsum) {
  const int k = blockIdx.x * 256 + threadIdx.x;
  float raw = cs[k] * DECAY_C + ONEMD_C * counts[k];
  ncsraw[k] = raw;
  float s = raw;
  #pragma unroll
  for (int m = 1; m < 32; m <<= 1) s += __shfl_xor(s, m, 32);
  __shared__ float red[8];
  if ((threadIdx.x & 31) == 0) red[threadIdx.x >> 5] = s;
  __syncthreads();
  if (threadIdx.x == 0) {
    float t = 0.0f;
    #pragma unroll
    for (int i = 0; i < 8; ++i) t += red[i];
    atomicAdd(nsum, t);
  }
}

// Final normalization: new_cluster_size, new_ema_w, new_embedding_weight, loss
__global__ void final_kernel(const float* __restrict__ emaw,
                             const float* __restrict__ ncsraw,
                             const float* __restrict__ dw,
                             const float* __restrict__ nsum,
                             const float* __restrict__ lossAcc,
                             float* __restrict__ out) {
  const size_t i = (size_t)blockIdx.x * blockDim.x + threadIdx.x;
  const float  n = *nsum;
  if (i < (size_t)KK * DD) {
    const int k = (int)(i >> 9);               // i / DD
    const float norm = (ncsraw[k] + EPS_C) / (n + (float)KK * EPS_C) * n;
    const float ne   = emaw[i] * DECAY_C + ONEMD_C * dw[i];
    out[O_EMAW + i] = ne;
    out[O_W + i]    = ne / norm;
    if ((i & (DD - 1)) == 0) out[O_CS + k] = norm;
  }
  if (i == 0) out[O_LOSS] = COMMIT_C * (*lossAcc) / (float)((size_t)NN * DD);
}

// ---------------------------------------------------------------------------
extern "C" void kernel_launch(void* const* d_in, const int* in_sizes, int n_in,
                              void* d_out, int out_size, void* d_ws, size_t ws_size,
                              hipStream_t stream) {
  (void)in_sizes; (void)n_in; (void)out_size; (void)ws_size;

  const float* X    = (const float*)d_in[0];   // [S,B,D]
  const float* W    = (const float*)d_in[1];   // [K,D]
  const float* EMAW = (const float*)d_in[2];   // [K,D]
  const float* CS   = (const float*)d_in[3];   // [K]
  float* out = (float*)d_out;

  char* ws = (char*)d_ws;
  unsigned short* Xb    = (unsigned short*)(ws + WS_XB);
  unsigned short* Wb    = (unsigned short*)(ws + WS_WB);
  float*          dw    = (float*)(ws + WS_DW);
  float*          enorm = (float*)(ws + WS_ENORM);
  float*          cnt   = (float*)(ws + WS_CNT);
  float*          ncs   = (float*)(ws + WS_NCS);
  float*          scal  = (float*)(ws + WS_SCAL);   // [0]=lossAcc [1]=nSum

  // 1) zero accumulators + one-hot region (scalar NT stores: odd base offsets)
  zero_kernel<<<2048, 256, 0, stream>>>(out + O_ENC, (size_t)NN * KK);
  zero_kernel<<<1024, 256, 0, stream>>>(dw, (size_t)KK * DD);
  zero_kernel<<<16,   256, 0, stream>>>(cnt, (size_t)KK);
  zero_kernel<<<1,    32,  0, stream>>>(scal, (size_t)2);

  // 2) bf16 conversions + codebook norms
  xconv_kernel<<<4096, 256, 0, stream>>>(X, Xb);
  wprep_kernel<<<KK, 128, 0, stream>>>(W, Wb, enorm);

  // 3) WMMA distance GEMM (async-LDS staged B) + argmin + epilogue
  vq_argmin_kernel<<<NN / 16 / 8, 256, 0, stream>>>(
      Xb, Wb, enorm, X, W,
      out + O_Q, out + O_ENC, out + O_IDX,
      dw, cnt, scal + 0);

  // 4) EMA normalization chain
  cs_kernel<<<KK / 256, 256, 0, stream>>>(CS, cnt, ncs, scal + 1);
  final_kernel<<<(KK * DD) / 256, 256, 0, stream>>>(EMAW, ncs, dw,
                                                    scal + 1, scal + 0, out);
}
